// AutoEncoderTopK_64656437674497
// MI455X (gfx1250) — matrix operand order repaired and checked by reading, not compiled, not run on hardware
//
#include <hip/hip_runtime.h>

#define DIM_B  8192
#define DIM_DA 4096
#define DIM_DD 16384
#define TOPK   64

typedef __attribute__((ext_vector_type(16))) __bf16 v16bf;
typedef __attribute__((ext_vector_type(8)))  float  v8f;

union FragU { uint4 q[2]; v16bf v; };

__device__ __forceinline__ unsigned short f2bf(float x) {
  // round-to-nearest-even f32 -> bf16
  unsigned u = __float_as_uint(x);
  unsigned r = u + 0x7FFFu + ((u >> 16) & 1u);
  return (unsigned short)(r >> 16);
}

// C[m,n] = act( sum_k (A[m,k] - center[k]) * Bm[n,k] + bias[n] )
// A: row-major [M,Kd] (lda), Bm: row-major [N,Kd] (ldb)  -- "NT" GEMM
// 1D grid, L2-friendly swizzle: groups of `gy` m-panels sweep all n-panels.
template <bool RELU, bool CENTER>
__global__ __launch_bounds__(256) void gemm_bf16_wmma(
    const float* __restrict__ A, int lda,
    const float* __restrict__ Bm, int ldb,
    const float* __restrict__ bias,
    const float* __restrict__ center,
    float* __restrict__ C, int N, int Kd, int gy) {
  constexpr int BM = 128, BN = 128, BK = 32;
  constexpr int PKA = 40;   // halves pitch of As rows (rows 16B-aligned, padded)
  constexpr int PNB = 136;  // halves pitch of Bs rows (rows 16B-aligned, padded)

  __shared__ __align__(16) unsigned short As[2][BM * PKA]; // [m][k] bf16
  __shared__ __align__(16) unsigned short Bs[2][BK * PNB]; // [k][n] bf16 (transposed)

  // ---- block swizzle for L2 reuse ----
  const int gridN = N / BN;
  const int tilesPerGroup = gy * gridN;
  const int g  = (int)blockIdx.x / tilesPerGroup;
  const int rr = (int)blockIdx.x % tilesPerGroup;
  const int m0 = (g * gy + (rr % gy)) * BM;
  const int n0 = (rr / gy) * BN;

  const int t = threadIdx.x;
  const int lane = t & 31;
  const int wave = t >> 5;          // 0..7
  const int wm = wave & 3;          // row group: 4 x 32 rows
  const int wn = wave >> 2;         // col group: 2 x 64 cols

  // A loader: thread -> (row lm, 16 consecutive k at lk)
  const int lm = t >> 1;
  const int lk = (t & 1) * 16;
  // W loader: thread -> (rows wn2, wn2+1 ; 8 consecutive k at wkg)
  const int wn2 = (t >> 2) * 2;     // 0,2,...,126
  const int wkg = (t & 3) * 8;      // 0,8,16,24

  const int r  = lane & 15;
  const int hi = lane >> 4;

  v8f zero = {0.f, 0.f, 0.f, 0.f, 0.f, 0.f, 0.f, 0.f};
  v8f acc[2][4];
#pragma unroll
  for (int ti = 0; ti < 2; ++ti)
#pragma unroll
    for (int tj = 0; tj < 4; ++tj) acc[ti][tj] = zero;

  const float* Arow  = A  + (size_t)(m0 + lm) * lda + lk;
  const float* Wrow0 = Bm + (size_t)(n0 + wn2) * ldb + wkg;
  const float* Wrow1 = Wrow0 + ldb;

  // -------- staging helpers --------
  auto loadStage = [&](int buf, int k0) {
    float4 av[4], w0[2], w1[2];
    {
      const float4* ap  = reinterpret_cast<const float4*>(Arow + k0);
      const float4* wp0 = reinterpret_cast<const float4*>(Wrow0 + k0);
      const float4* wp1 = reinterpret_cast<const float4*>(Wrow1 + k0);
#pragma unroll
      for (int j = 0; j < 4; ++j) av[j] = ap[j];
#pragma unroll
      for (int j = 0; j < 2; ++j) { w0[j] = wp0[j]; w1[j] = wp1[j]; }
    }
    // A: center + cvt, two 16B stores
    {
      const float* avf = reinterpret_cast<const float*>(av);
      __align__(16) unsigned short tmpA[16];
#pragma unroll
      for (int i = 0; i < 16; ++i) {
        float x = avf[i];
        if (CENTER) x -= center[k0 + lk + i];
        tmpA[i] = f2bf(x);
      }
      unsigned short* dst = &As[buf][lm * PKA + lk];
      *reinterpret_cast<uint4*>(dst)     = *reinterpret_cast<uint4*>(&tmpA[0]);
      *reinterpret_cast<uint4*>(dst + 8) = *reinterpret_cast<uint4*>(&tmpA[8]);
    }
    // W: transpose, packed 2xbf16 b32 stores
    {
      const float* w0f = reinterpret_cast<const float*>(w0);
      const float* w1f = reinterpret_cast<const float*>(w1);
#pragma unroll
      for (int i = 0; i < 8; ++i) {
        unsigned pk = (unsigned)f2bf(w0f[i]) | ((unsigned)f2bf(w1f[i]) << 16);
        *reinterpret_cast<unsigned*>(&Bs[buf][(wkg + i) * PNB + wn2]) = pk;
      }
    }
  };

  auto mmaStep = [&](int buf) {
    FragU af[2], bfg[4];
#pragma unroll
    for (int ti = 0; ti < 2; ++ti) {
      const unsigned short* p = &As[buf][(wm * 32 + ti * 16 + r) * PKA + hi * 8];
      af[ti].q[0] = *reinterpret_cast<const uint4*>(p);
      af[ti].q[1] = *reinterpret_cast<const uint4*>(p + 16);
    }
#pragma unroll
    for (int tj = 0; tj < 4; ++tj) {
      const unsigned short* p = &Bs[buf][(hi * 8) * PNB + wn * 64 + tj * 16 + r];
      unsigned w[8];
#pragma unroll
      for (int j = 0; j < 8; ++j) {
        const int kk = (j < 4) ? 2 * j : 16 + 2 * (j - 4);
        w[j] = (unsigned)p[kk * PNB] | ((unsigned)p[(kk + 1) * PNB] << 16);
      }
      bfg[tj].q[0] = make_uint4(w[0], w[1], w[2], w[3]);
      bfg[tj].q[1] = make_uint4(w[4], w[5], w[6], w[7]);
    }
#pragma unroll
    for (int ti = 0; ti < 2; ++ti)
#pragma unroll
      for (int tj = 0; tj < 4; ++tj)
        acc[ti][tj] = __builtin_amdgcn_wmma_f32_16x16x32_bf16(
            false, af[ti].v, false, bfg[tj].v, (short)0, acc[ti][tj], false, false);
  };

  // -------- double-buffered main loop --------
  const int nIter = Kd / BK;       // even for all our shapes
  loadStage(0, 0);
  __syncthreads();

  auto step = [&](int cur, int k0next, bool more) {
    if (more) loadStage(cur ^ 1, k0next);   // prefetch overlaps with WMMA below
    mmaStep(cur);
    __syncthreads();
  };
  for (int it = 0; it < nIter; it += 2) {
    step(0, (it + 1) * BK, it + 1 < nIter);
    step(1, (it + 2) * BK, it + 2 < nIter);
  }

  // -------- epilogue: bias + activation --------
#pragma unroll
  for (int ti = 0; ti < 2; ++ti) {
    const int mbase = m0 + wm * 32 + ti * 16 + hi * 8;
#pragma unroll
    for (int tj = 0; tj < 4; ++tj) {
      const int col = n0 + wn * 64 + tj * 16 + r;
      const float b = bias[col];
#pragma unroll
      for (int v = 0; v < 8; ++v) {
        float val = acc[ti][tj][v] + b;
        if (RELU) val = fmaxf(val, 0.0f);
        C[(size_t)(mbase + v) * N + col] = val;
      }
    }
  }
}

// Per-row exact top-K selection, zeroing all non-selected entries in place.
// Radix-select on float bits (post-ReLU => unsigned order == float order).
// Deterministic tie-break via LDS prefix scan.
__global__ __launch_bounds__(256) void topk_inplace(float* __restrict__ F) {
  extern __shared__ unsigned smem[];
  unsigned* vals = smem;            // DIM_DD values
  unsigned* hist = smem + DIM_DD;   // 256 bins (reused for scan)
  __shared__ unsigned s_digit, s_above;

  const int t = threadIdx.x;
  float* row = F + (size_t)blockIdx.x * DIM_DD;

  for (int i = t; i < DIM_DD; i += 256) vals[i] = __float_as_uint(row[i]);

  unsigned pmask = 0u, pfx = 0u;
  unsigned want = TOPK;

  for (int pass = 0; pass < 4; ++pass) {
    const int shift = 24 - 8 * pass;
    hist[t] = 0u;
    __syncthreads();
    for (int i = t; i < DIM_DD; i += 256) {
      unsigned u = vals[i];
      if ((u & pmask) == pfx) atomicAdd(&hist[(u >> shift) & 255u], 1u);
    }
    __syncthreads();
    if (t == 0) {
      unsigned run = 0u;
      int dsel = 0;
      for (int d = 255; d >= 0; --d) {
        run += hist[d];
        if (run >= want) { dsel = d; break; }
      }
      s_digit = (unsigned)dsel;
      s_above = run - hist[dsel];
    }
    __syncthreads();
    unsigned dsel = s_digit;
    want -= s_above;
    pfx |= dsel << shift;
    pmask |= 0xFFu << shift;
    __syncthreads();
  }

  const unsigned thresh = pfx;      // exact bit pattern of K-th largest

  unsigned myEq = 0u;
  for (int i = t; i < DIM_DD; i += 256)
    if (vals[i] == thresh) myEq++;

  hist[t] = myEq;
  __syncthreads();
#pragma unroll
  for (int off = 1; off < 256; off <<= 1) {
    unsigned v = (t >= off) ? hist[t - off] : 0u;
    __syncthreads();
    hist[t] += v;
    __syncthreads();
  }
  const unsigned base = hist[t] - myEq;

  unsigned local = 0u;
  for (int i = t; i < DIM_DD; i += 256) {
    unsigned u = vals[i];
    bool keep = (u > thresh);
    if (u == thresh) { keep = (base + local) < want; local++; }
    row[i] = keep ? __uint_as_float(u) : 0.0f;
  }
}

extern "C" void kernel_launch(void* const* d_in, const int* in_sizes, int n_in,
                              void* d_out, int out_size, void* d_ws, size_t ws_size,
                              hipStream_t stream) {
  (void)in_sizes; (void)n_in; (void)out_size; (void)d_ws; (void)ws_size;
  const float* x    = (const float*)d_in[0];   // [B, DA]
  const float* Wenc = (const float*)d_in[1];   // [DD, DA]
  const float* benc = (const float*)d_in[2];   // [DD]
  const float* Wdec = (const float*)d_in[3];   // [DA, DD]
  const float* bdec = (const float*)d_in[4];   // [DA]

  float* xhat = (float*)d_out;                       // [B, DA]
  float* f    = xhat + (size_t)DIM_B * DIM_DA;       // [B, DD]

  dim3 blk(256);
  const int gridM = DIM_B / 128;                     // 64

  // Encode: f = relu((x - b_dec) @ W_enc^T + b_enc)
  {
    const int gridN = DIM_DD / 128;                  // 128
    gemm_bf16_wmma<true, true><<<dim3(gridM * gridN), blk, 0, stream>>>(
        x, DIM_DA, Wenc, DIM_DA, benc, bdec, f, DIM_DD, DIM_DA, /*gy=*/16);
  }

  // Top-K per row, zero the rest (in place)
  size_t sm = (size_t)(DIM_DD + 256) * sizeof(unsigned);
  topk_inplace<<<dim3(DIM_B), blk, sm, stream>>>(f);

  // Decode: x_hat = f @ W_dec^T + b_dec (dense over full dict; f is sparse-zeroed)
  {
    const int gridN = DIM_DA / 128;                  // 32
    gemm_bf16_wmma<false, false><<<dim3(gridM * gridN), blk, 0, stream>>>(
        f, DIM_DD, Wdec, DIM_DD, bdec, nullptr, xhat, DIM_DA, DIM_DD, /*gy=*/8);
  }
}
